// NTM_10642928960211
// MI455X (gfx1250) — matrix operand order repaired and checked by reading, not compiled, not run on hardware
//
#include <hip/hip_runtime.h>

// Problem constants (from reference)
#define TT 64
#define BB 256
#define NN 256
#define MM 128
#define HH 1024
#define II 128
#define LDP 576   // padded fused-head output width (524 -> 576 = 9*64)
#define EPS 1e-8f

typedef __attribute__((ext_vector_type(16))) _Float16 v16h;
typedef __attribute__((ext_vector_type(8)))  _Float16 v8h;
typedef __attribute__((ext_vector_type(8)))  float    v8f;

// ---------------- helpers ----------------

__device__ __forceinline__ float sigmoidf_(float x) { return 1.0f / (1.0f + __expf(-x)); }
__device__ __forceinline__ float softplusf_(float x) { return (x > 20.f) ? x : log1pf(expf(x)); }

__device__ __forceinline__ float wave_reduce_sum(float v) {
#pragma unroll
  for (int m = 16; m > 0; m >>= 1) v += __shfl_xor(v, m, 32);
  return v;
}

// 256-thread (8-wave) block reductions; red must be float[8] shared
__device__ __forceinline__ float block_reduce_max256(float v, float* red) {
#pragma unroll
  for (int m = 16; m > 0; m >>= 1) v = fmaxf(v, __shfl_xor(v, m, 32));
  int wv = threadIdx.x >> 5;
  if ((threadIdx.x & 31) == 0) red[wv] = v;
  __syncthreads();
  float r = red[0];
#pragma unroll
  for (int i = 1; i < 8; i++) r = fmaxf(r, red[i]);
  __syncthreads();
  return r;
}

__device__ __forceinline__ float block_reduce_sum256(float v, float* red) {
  v = wave_reduce_sum(v);
  int wv = threadIdx.x >> 5;
  if ((threadIdx.x & 31) == 0) red[wv] = v;
  __syncthreads();
  float r = 0.f;
#pragma unroll
  for (int i = 0; i < 8; i++) r += red[i];
  __syncthreads();
  return r;
}

// ---------------- WMMA fragment handling ----------------
// A 16x32 f16 (row-major source): lane l (r=l&15, h=l>>4) holds row r,
// K = {h*16..h*16+15} -> two contiguous 128-bit loads.
__device__ __forceinline__ v16h load_a_frag(const _Float16* __restrict__ A, int lda,
                                            int row0, int k0, int r, int h) {
  const _Float16* p = A + (size_t)(row0 + r) * lda + k0 + h * 16;
  v8h lo = *(const v8h*)(p);
  v8h hi = *(const v8h*)(p + 8);
  return __builtin_shufflevector(lo, hi, 0, 1, 2, 3, 4, 5, 6, 7,
                                 8, 9, 10, 11, 12, 13, 14, 15);
}

// Packed B layout: fragment (k_tile = k0/32, col_tile = c0/16) stored as
//   Wp[ ((col_tile*KT + k_tile)*32 + lane)*16 + e ],  e = 0..15
// so each lane reads its 16 elements as two contiguous 128-bit loads.
__device__ __forceinline__ v16h load_b_frag_packed(const _Float16* __restrict__ Wp,
                                                   int KT, int k0, int col0, int lane) {
  const _Float16* p = Wp + ((size_t)((col0 >> 4) * KT + (k0 >> 5)) * 32 + lane) * 16;
  v8h lo = *(const v8h*)(p);
  v8h hi = *(const v8h*)(p + 8);
  return __builtin_shufflevector(lo, hi, 0, 1, 2, 3, 4, 5, 6, 7,
                                 8, 9, 10, 11, 12, 13, 14, 15);
}

// packed index for element (k, c) of a [K x N] matrix in B-fragment layout
__device__ __forceinline__ size_t packed_b_index(int k, int c, int K) {
  int k_tile = k >> 5, kr = k & 31;
  int h = kr >> 4, e = kr & 15;
  int lane = h * 16 + (c & 15);
  return ((size_t)((c >> 4) * (K >> 5) + k_tile) * 32 + lane) * 16 + e;
}

// Branch-free k-segment: accumulate A[row0.., kbeg..kend) x W[kbeg..kend, col0..]
// A local k starts at 0 for this segment (aoff subtracted).
__device__ __forceinline__ void gemm_segment(const _Float16* __restrict__ A, int lda,
                                             int kbeg, int kend,
                                             const _Float16* __restrict__ Wp, int KT,
                                             int row0, int col0, int r, int h, int lane,
                                             v8f c[4]) {
#pragma unroll 2
  for (int k0 = kbeg; k0 < kend; k0 += 32) {
    v16h a = load_a_frag(A, lda, row0, k0 - kbeg, r, h);
#pragma unroll
    for (int j = 0; j < 4; j++) {
      v16h b = load_b_frag_packed(Wp, KT, k0, col0 + j * 16, lane);
      c[j] = __builtin_amdgcn_wmma_f32_16x16x32_f16(
          /*neg_a=*/false, a, /*neg_b=*/false, b,
          /*c_mod=*/(short)0, c[j], /*reuse_a=*/false, /*reuse_b=*/false);
    }
  }
}

// ---------------- WMMA GEMM: out[M x N] = act(A[M x K] @ W[K x N] + bias) ----------------
// A row-major, optionally split at `ksplit` between two sources (x_t part | rd part).
// W pre-packed in B-fragment layout. mode 0: f32 out. mode 1: tanh + f16 out.
// Block = 128 threads (4 waves). Wave tile 16x64, block tile 64x64.
__global__ void __launch_bounds__(128)
wmma_gemm_kernel(const _Float16* __restrict__ A0, int lda0,
                 const _Float16* __restrict__ A1, int lda1, int ksplit,
                 const _Float16* __restrict__ Wp, int ldw,
                 const float* __restrict__ bias,
                 float* __restrict__ outF, _Float16* __restrict__ outH,
                 int K, int mode) {
  const int lane = threadIdx.x & 31;
  const int wv   = threadIdx.x >> 5;
  const int r    = lane & 15;
  const int h    = lane >> 4;
  const int row0 = blockIdx.y * 64 + wv * 16;
  const int col0 = blockIdx.x * 64;
  const int KT   = K >> 5;
  const int ks   = (ksplit < K) ? ksplit : K;

  v8f c[4] = {};  // 4 accumulators: 16x64 per wave

  gemm_segment(A0, lda0, 0, ks, Wp, KT, row0, col0, r, h, lane, c);
  gemm_segment(A1, lda1, ks, K, Wp, KT, row0, col0, r, h, lane, c);

  // Epilogue. C/D layout: VGPR v -> row = row0 + v + h*8, col = col0 + j*16 + r
#pragma unroll
  for (int j = 0; j < 4; j++) {
#pragma unroll
    for (int v = 0; v < 8; v++) {
      int row = row0 + v + h * 8;
      int col = col0 + j * 16 + r;
      float val = c[j][v] + bias[col];
      if (mode == 1) outH[(size_t)row * ldw + col] = (_Float16)tanhf(val);
      else           outF[(size_t)row * ldw + col] = val;
    }
  }
}

// ---------------- Content addressing scores ----------------
// One block per batch b (256 thr). Computes cos(k_i, mem[b,n,:]) for both heads.
__global__ void __launch_bounds__(256)
content_scores_kernel(const float* __restrict__ preact, const float* __restrict__ mem,
                      float* __restrict__ score0, float* __restrict__ score1) {
  const int b   = blockIdx.x;
  const int tid = threadIdx.x;
  __shared__ float k0s[MM], k1s[MM];
  __shared__ float knorm[2];

  if (tid < MM) {
    float v0 = preact[(size_t)b * LDP + tid];
    float v1 = preact[(size_t)b * LDP + MM + tid];
    k0s[tid] = v0 > 0.f ? v0 : 0.f;   // relu(k)
    k1s[tid] = v1 > 0.f ? v1 : 0.f;
  }
  __syncthreads();

  if (tid < 32) {
    float s = 0.f;
#pragma unroll
    for (int j = 0; j < 4; j++) { float v = k0s[tid + 32 * j]; s += v * v; }
    s = wave_reduce_sum(s);
    if (tid == 0) knorm[0] = fmaxf(sqrtf(s), EPS);
  } else if (tid < 64) {
    int l = tid - 32;
    float s = 0.f;
#pragma unroll
    for (int j = 0; j < 4; j++) { float v = k1s[l + 32 * j]; s += v * v; }
    s = wave_reduce_sum(s);
    if ((tid & 31) == 0) knorm[1] = fmaxf(sqrtf(s), EPS);
  }
  __syncthreads();

  const int wv = tid >> 5, lane = tid & 31;
  for (int i = 0; i < 32; i++) {
    int n = wv * 32 + i;
    const float4 m4 = *(const float4*)(mem + ((size_t)(b * NN + n)) * MM + lane * 4);
    float d0 = m4.x * k0s[lane * 4] + m4.y * k0s[lane * 4 + 1] +
               m4.z * k0s[lane * 4 + 2] + m4.w * k0s[lane * 4 + 3];
    float d1 = m4.x * k1s[lane * 4] + m4.y * k1s[lane * 4 + 1] +
               m4.z * k1s[lane * 4 + 2] + m4.w * k1s[lane * 4 + 3];
    float mm = m4.x * m4.x + m4.y * m4.y + m4.z * m4.z + m4.w * m4.w;
    d0 = wave_reduce_sum(d0);
    d1 = wave_reduce_sum(d1);
    mm = wave_reduce_sum(mm);
    if (lane == 0) {
      float mn = fmaxf(sqrtf(mm), EPS);
      score0[b * NN + n] = d0 / (knorm[0] * mn);
      score1[b * NN + n] = d1 / (knorm[1] * mn);
    }
  }
}

// ---------------- Addressing: softmax / interpolate / shift / sharpen ----------------
// grid (B, 2); 256 threads; thread per memory slot n.
__global__ void __launch_bounds__(256)
addressing_kernel(const float* __restrict__ preact,
                  const float* __restrict__ score0, const float* __restrict__ score1,
                  float* __restrict__ ww, float* __restrict__ rw) {
  const int b = blockIdx.x, hd = blockIdx.y, n = threadIdx.x;
  __shared__ float wbuf[NN];
  __shared__ float red[8];

  const float* pa = preact + (size_t)b * LDP + 512 + hd * 6;
  const float beta  = softplusf_(pa[0]);
  const float g     = sigmoidf_(pa[1]);
  float s0 = pa[2], s1 = pa[3], s2 = pa[4];
  const float gamma = fmaxf(pa[5], 0.f) + 1.0f;

  // shift softmax over S=3 (computed redundantly per thread)
  float sm = fmaxf(s0, fmaxf(s1, s2));
  float e0 = expf(s0 - sm), e1 = expf(s1 - sm), e2 = expf(s2 - sm);
  float es = e0 + e1 + e2;
  e0 /= es; e1 /= es; e2 /= es;

  const float* sc = hd ? score1 : score0;
  float* wio = hd ? rw : ww;

  float a = beta * sc[b * NN + n];
  float amax = block_reduce_max256(a, red);
  float ex = expf(a - amax);
  float ssum = block_reduce_sum256(ex, red);
  float w = ex / ssum;

  float prev = wio[b * NN + n];
  w = g * w + (1.0f - g) * prev;
  wbuf[n] = w;
  __syncthreads();

  // circular conv shift (S=3): w'[n] = e0*w[n-1] + e1*w[n] + e2*w[n+1]
  float shifted = e0 * wbuf[(n + NN - 1) & (NN - 1)] + e1 * wbuf[n] +
                  e2 * wbuf[(n + 1) & (NN - 1)];
  float sharp = powf(shifted, gamma);
  float tot = block_reduce_sum256(sharp, red);
  wio[b * NN + n] = sharp / tot;
}

// ---------------- Memory update + read + output projection ----------------
// One block (128 thr) per batch b; thread per memory column m.
__global__ void __launch_bounds__(128)
update_read_out_kernel(const float* __restrict__ preact,
                       const float* __restrict__ ww, const float* __restrict__ rw,
                       float* __restrict__ mem, _Float16* __restrict__ rd_f16,
                       const float* __restrict__ Wo, const float* __restrict__ bo,
                       float* __restrict__ out_t) {
  const int b = blockIdx.x, m = threadIdx.x;
  __shared__ float rds[MM];

  const float e = sigmoidf_(preact[(size_t)b * LDP + 256 + m]);
  const float a = sigmoidf_(preact[(size_t)b * LDP + 384 + m]);

  float acc = 0.f;
  size_t base = (size_t)b * NN * MM + m;
  for (int n = 0; n < NN; n++) {
    float w = ww[b * NN + n];
    float r = rw[b * NN + n];
    float mv = mem[base + (size_t)n * MM];
    mv = mv * (1.0f - w * e) + w * a;
    mem[base + (size_t)n * MM] = mv;
    acc += r * mv;
  }
  rds[m] = acc;
  rd_f16[b * MM + m] = (_Float16)acc;
  __syncthreads();

  float o = bo[m];
  for (int k = 0; k < MM; k++) o += rds[k] * Wo[k * II + m];
  out_t[(size_t)b * II + m] = o;
}

// ---------------- Setup kernels ----------------

__global__ void cvt_f32_to_f16_kernel(_Float16* __restrict__ dst,
                                      const float* __restrict__ src, int n) {
  for (int i = blockIdx.x * blockDim.x + threadIdx.x; i < n; i += gridDim.x * blockDim.x)
    dst[i] = (_Float16)src[i];
}

// Pack a row-major [K x N] f32 matrix into f16 B-fragment layout.
__global__ void pack_b_kernel(const float* __restrict__ src, _Float16* __restrict__ dst,
                              int K, int N) {
  int idx = blockIdx.x * blockDim.x + threadIdx.x;
  if (idx >= K * N) return;
  int k = idx / N, c = idx % N;
  dst[packed_b_index(k, c, K)] = (_Float16)src[idx];
}

// Build fused head weight matrix [H x LDP] directly into packed B-fragment layout:
// columns [k0|k1|erase|add|scalars|pad].
__global__ void build_whead_kernel(const float* __restrict__ Wk, const float* __restrict__ We,
                                   const float* __restrict__ Wa, const float* __restrict__ Wb,
                                   const float* __restrict__ Wg, const float* __restrict__ Ws,
                                   const float* __restrict__ Wgam, _Float16* __restrict__ out) {
  int idx = blockIdx.x * blockDim.x + threadIdx.x;
  if (idx >= HH * LDP) return;
  int r = idx / LDP, c = idx % LDP;
  float v = 0.f;
  if (c < 256) {
    int hd = c >> 7, m = c & 127;
    v = Wk[(size_t)hd * HH * MM + (size_t)r * MM + m];
  } else if (c < 384) {
    v = We[(size_t)r * MM + (c - 256)];
  } else if (c < 512) {
    v = Wa[(size_t)r * MM + (c - 384)];
  } else if (c < 524) {
    int hd = (c - 512) / 6, o = (c - 512) % 6;
    if (o == 0)      v = Wb[hd * HH + r];
    else if (o == 1) v = Wg[hd * HH + r];
    else if (o <= 4) v = Ws[(size_t)(hd * HH + r) * 3 + (o - 2)];
    else             v = Wgam[hd * HH + r];
  }
  out[packed_b_index(r, c, HH)] = (_Float16)v;
}

__global__ void build_bias_kernel(const float* __restrict__ bk, const float* __restrict__ be,
                                  const float* __restrict__ ba, const float* __restrict__ bb,
                                  const float* __restrict__ bg, const float* __restrict__ bs,
                                  const float* __restrict__ bgam, float* __restrict__ out) {
  int c = blockIdx.x * blockDim.x + threadIdx.x;
  if (c >= LDP) return;
  float v = 0.f;
  if (c < 256)      v = bk[c];
  else if (c < 384) v = be[c - 256];
  else if (c < 512) v = ba[c - 384];
  else if (c < 524) {
    int hd = (c - 512) / 6, o = (c - 512) % 6;
    if (o == 0)      v = bb[hd];
    else if (o == 1) v = bg[hd];
    else if (o <= 4) v = bs[hd * 3 + (o - 2)];
    else             v = bgam[hd];
  }
  out[c] = v;
}

__global__ void tile_memory_kernel(const float* __restrict__ m0, float* __restrict__ mem) {
  const int per = NN * MM;
  for (size_t i = blockIdx.x * blockDim.x + threadIdx.x; i < (size_t)BB * per;
       i += (size_t)gridDim.x * blockDim.x)
    mem[i] = m0[i % per];
}

// softmax over N=256 of a [1,N] vector, tiled to [B,N]
__global__ void __launch_bounds__(256)
softmax_tile_kernel(const float* __restrict__ w0, float* __restrict__ out) {
  __shared__ float red[8];
  int n = threadIdx.x;
  float v = w0[n];
  float mx = block_reduce_max256(v, red);
  float e = expf(v - mx);
  float s = block_reduce_sum256(e, red);
  float sm = e / s;
  for (int b = 0; b < BB; b++) out[b * NN + n] = sm;
}

__global__ void init_rd_kernel(const float* __restrict__ read0, _Float16* __restrict__ rd) {
  int i = blockIdx.x * blockDim.x + threadIdx.x;
  if (i < BB * MM) rd[i] = (_Float16)read0[i % MM];
}

// ---------------- host launch ----------------

extern "C" void kernel_launch(void* const* d_in, const int* in_sizes, int n_in,
                              void* d_out, int out_size, void* d_ws, size_t ws_size,
                              hipStream_t stream) {
  const float* x    = (const float*)d_in[0];
  const float* Wc   = (const float*)d_in[1];
  const float* bc   = (const float*)d_in[2];
  const float* Wk   = (const float*)d_in[3];
  const float* bk   = (const float*)d_in[4];
  const float* Wb   = (const float*)d_in[5];
  const float* bb   = (const float*)d_in[6];
  const float* Wg   = (const float*)d_in[7];
  const float* bg   = (const float*)d_in[8];
  const float* Ws   = (const float*)d_in[9];
  const float* bs   = (const float*)d_in[10];
  const float* Wgam = (const float*)d_in[11];
  const float* bgam = (const float*)d_in[12];
  const float* We   = (const float*)d_in[13];
  const float* be   = (const float*)d_in[14];
  const float* Wa   = (const float*)d_in[15];
  const float* ba   = (const float*)d_in[16];
  const float* Wo   = (const float*)d_in[17];
  const float* bo   = (const float*)d_in[18];
  const float* memory0 = (const float*)d_in[19];
  const float* ww0  = (const float*)d_in[20];
  const float* rw0  = (const float*)d_in[21];
  const float* read0 = (const float*)d_in[22];
  float* out = (float*)d_out;

  // workspace carve (256B aligned)
  char* p = (char*)d_ws;
  auto carve = [&](size_t bytes) -> void* {
    void* r = (void*)p;
    p += (bytes + 255) & ~(size_t)255;
    return r;
  };
  float*    mem       = (float*)carve((size_t)BB * NN * MM * 4);   // 32 MB
  float*    ww        = (float*)carve((size_t)BB * NN * 4);
  float*    rw        = (float*)carve((size_t)BB * NN * 4);
  float*    score0    = (float*)carve((size_t)BB * NN * 4);
  float*    score1    = (float*)carve((size_t)BB * NN * 4);
  _Float16* x_f16     = (_Float16*)carve((size_t)TT * BB * II * 2);
  _Float16* Wc_pk     = (_Float16*)carve((size_t)(II + MM) * HH * 2);
  _Float16* Whead_pk  = (_Float16*)carve((size_t)HH * LDP * 2);
  float*    bias_head = (float*)carve((size_t)LDP * 4);
  _Float16* ht_f16    = (_Float16*)carve((size_t)BB * HH * 2);
  float*    preact    = (float*)carve((size_t)BB * LDP * 4);
  _Float16* rd_f16    = (_Float16*)carve((size_t)BB * MM * 2);

  // one-time setup (re-run every call: deterministic)
  cvt_f32_to_f16_kernel<<<1024, 256, 0, stream>>>(x_f16, x, TT * BB * II);
  pack_b_kernel<<<((II + MM) * HH + 255) / 256, 256, 0, stream>>>(Wc, Wc_pk, II + MM, HH);
  build_whead_kernel<<<(HH * LDP + 255) / 256, 256, 0, stream>>>(Wk, We, Wa, Wb, Wg, Ws, Wgam,
                                                                 Whead_pk);
  build_bias_kernel<<<(LDP + 127) / 128, 128, 0, stream>>>(bk, be, ba, bb, bg, bs, bgam,
                                                           bias_head);
  tile_memory_kernel<<<2048, 256, 0, stream>>>(memory0, mem);
  softmax_tile_kernel<<<1, 256, 0, stream>>>(ww0, ww);
  softmax_tile_kernel<<<1, 256, 0, stream>>>(rw0, rw);
  init_rd_kernel<<<(BB * MM + 255) / 256, 256, 0, stream>>>(read0, rd_f16);

  for (int t = 0; t < TT; t++) {
    // 1) controller: ht = tanh([x_t | rd] @ Wc + bc), f16 out
    wmma_gemm_kernel<<<dim3(HH / 64, BB / 64), 128, 0, stream>>>(
        x_f16 + (size_t)t * BB * II, II, rd_f16, MM, /*ksplit=*/II,
        Wc_pk, /*ldw=*/HH, bc, nullptr, ht_f16, /*K=*/II + MM, /*mode=*/1);
    // 2) fused head projection: preact = ht @ Whead + bias, f32 out
    wmma_gemm_kernel<<<dim3(LDP / 64, BB / 64), 128, 0, stream>>>(
        ht_f16, HH, ht_f16, HH, /*ksplit=*/HH,
        Whead_pk, /*ldw=*/LDP, bias_head, preact, nullptr, /*K=*/HH, /*mode=*/0);
    // 3) cosine-similarity content scores for both heads
    content_scores_kernel<<<BB, 256, 0, stream>>>(preact, mem, score0, score1);
    // 4) addressing (softmax, interpolate, circular shift, sharpen) per head
    addressing_kernel<<<dim3(BB, 2), 256, 0, stream>>>(preact, score0, score1, ww, rw);
    // 5) memory erase/add update + read + output projection
    update_read_out_kernel<<<BB, 128, 0, stream>>>(preact, ww, rw, mem, rd_f16, Wo, bo,
                                                   out + (size_t)t * BB * II);
  }
}
